// Rvt_71055938945643
// MI455X (gfx1250) — compile-verified, matrix operand order checked by reading, not run on hardware
//
#include <hip/hip_runtime.h>
#include <hip/hip_bf16.h>
#include <cstdint>

// ---------------------------------------------------------------------------
// MI455X (gfx1250) conv-LSTM-attention. All heavy math via
// v_wmma_f32_16x16x32_bf16 (wave32 WMMA, f32 accumulate).
// Fragment layouts per CDNA5 ISA 7.12.2: per lane, a fragment is two
// contiguous 8-element runs (K = ks*32 + lh*8 + [0..7] / +16+[0..7]),
// so with the right LDS layout every fragment is two 16B vector loads.
// BK=64: one LDS stage feeds 2 WMMA K-steps (8 WMMAs per barrier pair).
// ---------------------------------------------------------------------------

typedef __attribute__((ext_vector_type(16))) __bf16 v16bf;
typedef __attribute__((ext_vector_type(8)))  __bf16 v8bf;
typedef __attribute__((ext_vector_type(8)))  float  v8f;

static __device__ __forceinline__ v16bf cat16(v8bf a, v8bf b) {
    return __builtin_shufflevector(a, b, 0,1,2,3,4,5,6,7,8,9,10,11,12,13,14,15);
}

// Problem constants
#define NB   16
#define IC   128
#define RC   256
#define AC   256
#define HWD  24
#define QP   576          // 24*24
#define DV   484          // 22*22
#define KSM  4608         // 512*9
#define MSAME 1280        // 256 (q) + 1024 (gates)
#define MVAL  512         // 256 (k) + 256 (v)

// ---------------------------------------------------------------------------
// B-tile loaders: fill a column-major LDS tile Bs[col][k] (64 cols x 64 k).
// 128 threads: thread -> k-row r = tid>>1, column segment cseg = (tid&1)*32.
// All loads are unconditional (clamped addresses); OOB columns either get a
// cndmask'd zero (SAME pad) or harmless garbage that the store guard drops.
// ---------------------------------------------------------------------------
struct BDense {
    const __bf16* base; long bstride; int ld;
    __device__ __forceinline__ void fill(__bf16* Bs, int b, int k0, int n0, int tid) const {
        const int r = tid >> 1, cseg = (tid & 1) * 32;
        const __bf16* src = base + (size_t)b * bstride + (size_t)(k0 + r) * ld + n0 + cseg;
        v8bf v0 = *(const v8bf*)src;
        v8bf v1 = *(const v8bf*)(src + 8);
        v8bf v2 = *(const v8bf*)(src + 16);
        v8bf v3 = *(const v8bf*)(src + 24);
#pragma unroll
        for (int j = 0; j < 8; ++j) {
            Bs[(cseg + j)      * 64 + r] = v0[j];
            Bs[(cseg + 8 + j)  * 64 + r] = v1[j];
            Bs[(cseg + 16 + j) * 64 + r] = v2[j];
            Bs[(cseg + 24 + j) * 64 + r] = v3[j];
        }
    }
};

struct BConvSame {   // im2col on the fly, SAME 3x3 over [xin ; h0], 24x24
    const __bf16* xin; const __bf16* h0;
    __device__ __forceinline__ void fill(__bf16* Bs, int b, int k0, int n0, int tid) const {
        const int r = tid >> 1, cseg = (tid & 1) * 32;
        const int k = k0 + r;
        const int c = k / 9, t = k - c * 9;
        const int ky = t / 3, kx = t - ky * 3;
        const __bf16* src = (c < RC) ? (xin + ((size_t)b * RC + c) * QP)
                                     : (h0  + ((size_t)b * RC + (c - RC)) * QP);
        int col = n0 + cseg;
        int y = col / HWD, x = col - y * HWD;
#pragma unroll
        for (int j = 0; j < 32; ++j) {
            int sy = y + ky - 1, sx = x + kx - 1;
            bool ok = ((unsigned)sy < (unsigned)HWD) & ((unsigned)sx < (unsigned)HWD);
            __bf16 v = src[ok ? sy * HWD + sx : 0];      // unconditional load
            Bs[(cseg + j) * 64 + r] = ok ? v : (__bf16)0.f;
            if (++x == HWD) { x = 0; ++y; }
        }
    }
};

struct BConvValid {  // VALID 3x3 -> 22x22. cols >=484 produce clamped garbage
    const __bf16* xin; const __bf16* h0;  //   (dropped by the output guard).
    __device__ __forceinline__ void fill(__bf16* Bs, int b, int k0, int n0, int tid) const {
        const int r = tid >> 1, cseg = (tid & 1) * 32;
        const int k = k0 + r;
        const int c = k / 9, t = k - c * 9;
        const int ky = t / 3, kx = t - ky * 3;
        const __bf16* src = (c < RC) ? (xin + ((size_t)b * RC + c) * QP)
                                     : (h0  + ((size_t)b * RC + (c - RC)) * QP);
        int col = n0 + cseg;
        int y = col / 22, x = col - y * 22;
#pragma unroll
        for (int j = 0; j < 32; ++j) {
            int sy = y + ky; if (sy > 23) sy = 23;        // in-bounds for pad cols
            int sx = x + kx; if (sx > 23) sx = 23;
            Bs[(cseg + j) * 64 + r] = src[sy * HWD + sx];
            if (++x == 22) { x = 0; ++y; }
        }
    }
};

// ---------------------------------------------------------------------------
// Generic LDS-tiled bf16 WMMA GEMM: C[b] = A[M,K] * B[b][K,N] (+bias,+Cinit)
// 128 threads = 4 waves; 64x64 tile; BK=64 = two WMMA K-steps per stage.
// A LDS tile row-major [row][64]; B LDS tile column-major [col][64].
// M must be a multiple of 64 and K a multiple of 64 (true for all uses).
// ---------------------------------------------------------------------------
template <typename BL>
__global__ void __launch_bounds__(128)
gemm_wmma(const __bf16* __restrict__ A, BL bl,
          const float* __restrict__ Cinit, long cinit_bstride,
          const float* __restrict__ bias,
          float* __restrict__ Coutf, __bf16* __restrict__ Coutb,
          long cout_bstride, int M, int N, int K)
{
    __shared__ __bf16 As[64 * 64];
    __shared__ __bf16 Bs[64 * 64];
    const int b  = blockIdx.z;
    const int m0 = blockIdx.y * 64, n0 = blockIdx.x * 64;
    const int tid = threadIdx.x, wave = tid >> 5, lane = tid & 31;
    const int lm = lane & 15, lh = lane >> 4;
    const int wm = (wave >> 1) * 32, wn = (wave & 1) * 32;
    const int ar = tid >> 1, ah = (tid & 1) * 32;       // A-fill: row + 32-col half

    v8f acc[2][2] = {};
    for (int k0 = 0; k0 < K; k0 += 64) {
        {   // A tile: 2 threads/row, four 16B loads + four 16B LDS stores each
            const __bf16* src = A + (size_t)(m0 + ar) * K + k0 + ah;
            *(v8bf*)&As[ar * 64 + ah]      = *(const v8bf*)src;
            *(v8bf*)&As[ar * 64 + ah + 8]  = *(const v8bf*)(src + 8);
            *(v8bf*)&As[ar * 64 + ah + 16] = *(const v8bf*)(src + 16);
            *(v8bf*)&As[ar * 64 + ah + 24] = *(const v8bf*)(src + 24);
            if (k0 + 64 < K)                               // gfx1250 prefetch path
                __builtin_prefetch(src + 64, 0, 1);
        }
        bl.fill(Bs, b, k0, n0, tid);
        __syncthreads();

#pragma unroll
        for (int ks = 0; ks < 2; ++ks) {
            v16bf af[2], bfr[2];
#pragma unroll
            for (int im = 0; im < 2; ++im) {
                const __bf16* p = &As[(wm + im * 16 + lm) * 64 + ks * 32 + lh * 8];
                af[im] = cat16(*(const v8bf*)p, *(const v8bf*)(p + 16));
            }
#pragma unroll
            for (int in = 0; in < 2; ++in) {
                const __bf16* p = &Bs[(wn + in * 16 + lm) * 64 + ks * 32 + lh * 8];
                bfr[in] = cat16(*(const v8bf*)p, *(const v8bf*)(p + 16));
            }
#pragma unroll
            for (int im = 0; im < 2; ++im)
#pragma unroll
                for (int in = 0; in < 2; ++in)
                    acc[im][in] = __builtin_amdgcn_wmma_f32_16x16x32_bf16(
                        false, af[im], false, bfr[in], (short)0, acc[im][in], false, false);
        }
        __syncthreads();
    }

#pragma unroll
    for (int im = 0; im < 2; ++im)
#pragma unroll
        for (int in = 0; in < 2; ++in)
#pragma unroll
            for (int j = 0; j < 8; ++j) {
                int gm = m0 + wm + im * 16 + j + lh * 8;   // C/D: M = j + 8*laneHalf
                int gn = n0 + wn + in * 16 + lm;
                if (gn < N) {
                    float v = acc[im][in][j];
                    if (bias)  v += bias[gm];
                    if (Cinit) v += Cinit[(size_t)b * cinit_bstride + (size_t)gm * N + gn];
                    size_t idx = (size_t)b * cout_bstride + (size_t)gm * N + gn;
                    if (Coutf) Coutf[idx] = v;
                    if (Coutb) Coutb[idx] = (__bf16)v;
                }
            }
}

// ---------------------------------------------------------------------------
// Fused attention: per (batch*head, 16-row q stripe): scores (WMMA, K=HC=32)
// -> per-wave shfl softmax in LDS -> a = v * probs^T (WMMA, K 484->512 pad).
// kT layout [BH][484][32] (d-major) so B fragments are contiguous per lane.
// ---------------------------------------------------------------------------
__global__ void __launch_bounds__(128)
attn_fused(const __bf16* __restrict__ qT,   // [BH][576][32]
           const __bf16* __restrict__ kT,   // [BH][484][32]
           const __bf16* __restrict__ vb,   // [BH][32][512] (zero padded)
           __bf16* __restrict__ ab)         // [16][256][576]
{
    __shared__ float S[16][520];            // probs rows; 520 stride skews banks
    __shared__ float Pr[4][8][32];
    const int bh = blockIdx.y, qt = blockIdx.x;
    const int tid = threadIdx.x, wave = tid >> 5, lane = tid & 31;
    const int lm = lane & 15, lh = lane >> 4;

    // ---- stage 1: scores = q^T k --------------------------------------
    const __bf16* qp = qT + ((size_t)bh * QP + qt * 16 + lm) * 32 + lh * 8;
    v16bf afr = cat16(*(const v8bf*)qp, *(const v8bf*)(qp + 16));
    for (int ct = wave; ct < 31; ct += 4) {          // 31 col tiles cover 484
        int d = ct * 16 + lm; if (d > DV - 1) d = DV - 1;   // clamp; pad cols dead
        const __bf16* kp = kT + ((size_t)bh * DV + d) * 32 + lh * 8;
        v16bf bfr = cat16(*(const v8bf*)kp, *(const v8bf*)(kp + 16));
        v8f s = {};
        s = __builtin_amdgcn_wmma_f32_16x16x32_bf16(false, afr, false, bfr,
                                                    (short)0, s, false, false);
#pragma unroll
        for (int j = 0; j < 8; ++j)
            S[j + lh * 8][ct * 16 + lm] = s[j];
    }
    __syncthreads();

    // ---- stage 2: per-wave softmax (one row per wave, shfl reductions) ----
    for (int r = wave; r < 16; r += 4) {
        float mx = -3.4e38f;
        for (int d = lane; d < DV; d += 32) mx = fmaxf(mx, S[r][d]);
        for (int m = 16; m; m >>= 1) mx = fmaxf(mx, __shfl_xor(mx, m, 32));
        float sum = 0.f;
        for (int d = lane; d < DV; d += 32) { float e = __expf(S[r][d] - mx); S[r][d] = e; sum += e; }
        for (int m = 16; m; m >>= 1) sum += __shfl_xor(sum, m, 32);
        float inv = 1.f / sum;
        for (int d = lane; d < DV; d += 32) S[r][d] *= inv;
        for (int d = DV + lane; d < 520; d += 32) S[r][d] = 0.f;   // zero pad
    }
    __syncthreads();

    // ---- stage 3: a[:, qstripe] = v(32x512) * probs^T(512x16) -------------
    const int mtile = wave & 1, khalf = wave >> 1;    // waves split M and K
    const __bf16* vbase = vb + ((size_t)bh * 32 + mtile * 16 + lm) * 512;
    v8f acc = {};
    for (int k0 = khalf * 256; k0 < khalf * 256 + 256; k0 += 32) {
        const __bf16* vp = vbase + k0 + lh * 8;
        v16bf av = cat16(*(const v8bf*)vp, *(const v8bf*)(vp + 16));
        const float* sp = &S[lm][k0 + lh * 8];
        v8f lo = *(const v8f*)sp;
        v8f hi = *(const v8f*)(sp + 16);
        v16bf bw;
#pragma unroll
        for (int e = 0; e < 8; ++e) { bw[e] = (__bf16)lo[e]; bw[e + 8] = (__bf16)hi[e]; }
        acc = __builtin_amdgcn_wmma_f32_16x16x32_bf16(false, av, false, bw,
                                                      (short)0, acc, false, false);
    }
#pragma unroll
    for (int j = 0; j < 8; ++j) Pr[wave][j][lane] = acc[j];
    __syncthreads();
    if (tid < 64) {
        int w2 = tid >> 5, l2 = tid & 31, lm2 = l2 & 15, lh2 = l2 >> 4;
        int b = bh >> 3, head = bh & 7;
#pragma unroll
        for (int j = 0; j < 8; ++j) {
            float vsum = Pr[w2][j][l2] + Pr[w2 + 2][j][l2];
            int c  = w2 * 16 + j + lh2 * 8;
            int qq = qt * 16 + lm2;
            ab[((size_t)b * AC + head * 32 + c) * QP + qq] = (__bf16)vsum;
        }
    }
}

// ---------------------------------------------------------------------------
// Pointwise / packing kernels
// ---------------------------------------------------------------------------
__global__ void cast_bf16(const float* __restrict__ in, __bf16* __restrict__ out, size_t n) {
    size_t i = (size_t)blockIdx.x * blockDim.x + threadIdx.x;
    if (i < n) out[i] = (__bf16)in[i];
}

__global__ void pack_wsame(const float* wq_x, const float* wq_h,
                           const float* wg_x, const float* wg_h, __bf16* W) {
    size_t i = (size_t)blockIdx.x * blockDim.x + threadIdx.x;
    const size_t total = (size_t)MSAME * KSM;
    if (i >= total) return;
    int m = (int)(i / KSM), k = (int)(i % KSM);
    int c = k / 9, t = k % 9;
    float v;
    if (m < RC) v = (c < RC) ? wq_x[((size_t)m * RC + c) * 9 + t]
                             : wq_h[((size_t)m * RC + (c - RC)) * 9 + t];
    else { int g = m - RC;
           v = (c < RC) ? wg_x[((size_t)g * RC + c) * 9 + t]
                        : wg_h[((size_t)g * RC + (c - RC)) * 9 + t]; }
    W[i] = (__bf16)v;
}

__global__ void pack_wvalid(const float* wk_x, const float* wk_h,
                            const float* wv_x, const float* wv_h, __bf16* W) {
    size_t i = (size_t)blockIdx.x * blockDim.x + threadIdx.x;
    const size_t total = (size_t)MVAL * KSM;
    if (i >= total) return;
    int m = (int)(i / KSM), k = (int)(i % KSM);
    int c = k / 9, t = k % 9;
    float v;
    if (m < RC) v = (c < RC) ? wk_x[((size_t)m * RC + c) * 9 + t]
                             : wk_h[((size_t)m * RC + (c - RC)) * 9 + t];
    else { int g = m - RC;
           v = (c < RC) ? wv_x[((size_t)g * RC + c) * 9 + t]
                        : wv_h[((size_t)g * RC + (c - RC)) * 9 + t]; }
    W[i] = (__bf16)v;
}

__global__ void make_qT(const float* __restrict__ qg, __bf16* __restrict__ qT) {
    // qg: [16][1280][576] (rows 0..255 = q). qT: [BH][576][32]
    size_t i = (size_t)blockIdx.x * blockDim.x + threadIdx.x;
    const size_t total = (size_t)NB * 8 * QP * 32;
    if (i >= total) return;
    int c = (int)(i & 31); size_t t = i >> 5;
    int p = (int)(t % QP); t /= QP;
    int head = (int)(t & 7); int b = (int)(t >> 3);
    int ch = head * 32 + c;
    qT[i] = (__bf16)qg[((size_t)b * MSAME + ch) * QP + p];
}

__global__ void make_kv(const float* __restrict__ kvf, const float* __restrict__ bv,
                        __bf16* __restrict__ kT, __bf16* __restrict__ vb) {
    // kvf: [16][512][484] (rows 0..255=k, 256..511=v).
    // kT: [BH][484][32] (d-major)   vb: [BH][32][512] (zero padded)
    size_t i = (size_t)blockIdx.x * blockDim.x + threadIdx.x;
    const size_t total = (size_t)NB * 8 * 32 * 512;
    if (i >= total) return;
    int d = (int)(i & 511); size_t t = i >> 9;
    int c = (int)(t & 31); t >>= 5;
    int head = (int)(t & 7); int b = (int)(t >> 3);
    int ch = head * 32 + c;
    size_t bh = (size_t)b * 8 + head;
    float vv = (d < DV) ? kvf[((size_t)b * MVAL + RC + ch) * DV + d] + bv[ch] : 0.f;
    vb[(bh * 32 + c) * 512 + d] = (__bf16)vv;
    if (d < DV)
        kT[(bh * DV + d) * 32 + c] = (__bf16)kvf[((size_t)b * MVAL + ch) * DV + d];
}

__global__ void lstm_pw(const float* __restrict__ qg, const float* __restrict__ c0,
                        __bf16* __restrict__ hb) {
    size_t i = (size_t)blockIdx.x * blockDim.x + threadIdx.x;
    const size_t total = (size_t)NB * RC * QP;
    if (i >= total) return;
    int p = (int)(i % QP); size_t t = i / QP;
    int ch = (int)(t % RC); int b = (int)(t / RC);
    const float* gb = qg + (size_t)b * MSAME * QP + (size_t)RC * QP; // gates rows
    float iv = gb[(size_t)(0   + ch) * QP + p];
    float fv = gb[(size_t)(256 + ch) * QP + p];
    float gv = gb[(size_t)(512 + ch) * QP + p];
    float ov = gb[(size_t)(768 + ch) * QP + p];
    float si = 1.f / (1.f + __expf(-iv));
    float sf = 1.f / (1.f + __expf(-fv));
    float so = 1.f / (1.f + __expf(-ov));
    float c  = sf * c0[i] + si * tanhf(gv);
    float h  = so * tanhf(c);
    hb[i] = (__bf16)h;
}

// ---------------------------------------------------------------------------
// Host launcher
// ---------------------------------------------------------------------------
static inline size_t ws_alloc(size_t& off, size_t bytes) {
    off = (off + 255) & ~(size_t)255;
    size_t o = off; off += bytes; return o;
}

extern "C" void kernel_launch(void* const* d_in, const int* in_sizes, int n_in,
                              void* d_out, int out_size, void* d_ws, size_t ws_size,
                              hipStream_t stream) {
    (void)in_sizes; (void)n_in; (void)out_size; (void)ws_size;
    const float* x    = (const float*)d_in[0];
    const float* h0   = (const float*)d_in[1];
    const float* c0   = (const float*)d_in[2];
    const float* w_in = (const float*)d_in[3];
    const float* b_in = (const float*)d_in[4];
    const float* wq_x = (const float*)d_in[5];
    const float* wq_h = (const float*)d_in[6];
    const float* wk_x = (const float*)d_in[7];
    const float* wk_h = (const float*)d_in[8];
    const float* wv_x = (const float*)d_in[9];
    const float* wv_h = (const float*)d_in[10];
    const float* bv   = (const float*)d_in[11];
    const float* wg_a = (const float*)d_in[12];
    const float* bg   = (const float*)d_in[13];
    const float* wg_x = (const float*)d_in[14];
    const float* wg_h = (const float*)d_in[15];
    const float* w_out= (const float*)d_in[16];
    const float* b_out= (const float*)d_in[17];
    float* out = (float*)d_out;

    char* ws = (char*)d_ws;
    size_t off = 0;
    __bf16* xb     = (__bf16*)(ws + ws_alloc(off, (size_t)NB*IC*QP*2));
    __bf16* h0b    = (__bf16*)(ws + ws_alloc(off, (size_t)NB*RC*QP*2));
    __bf16* w_inb  = (__bf16*)(ws + ws_alloc(off, (size_t)RC*IC*2));
    __bf16* xinb   = (__bf16*)(ws + ws_alloc(off, (size_t)NB*RC*QP*2));
    __bf16* Wsame  = (__bf16*)(ws + ws_alloc(off, (size_t)MSAME*KSM*2));
    __bf16* Wvalid = (__bf16*)(ws + ws_alloc(off, (size_t)MVAL*KSM*2));
    float*  qg     = (float*) (ws + ws_alloc(off, (size_t)NB*MSAME*QP*4)); // q + gates
    float*  kvf    = (float*) (ws + ws_alloc(off, (size_t)NB*MVAL*DV*4));
    __bf16* qTb    = (__bf16*)(ws + ws_alloc(off, (size_t)NB*8*QP*32*2));
    __bf16* kTb    = (__bf16*)(ws + ws_alloc(off, (size_t)NB*8*DV*32*2));
    __bf16* vbb    = (__bf16*)(ws + ws_alloc(off, (size_t)NB*8*32*512*2));
    __bf16* ab     = (__bf16*)(ws + ws_alloc(off, (size_t)NB*AC*QP*2));
    __bf16* wgab   = (__bf16*)(ws + ws_alloc(off, (size_t)1024*RC*2));
    __bf16* hb     = (__bf16*)(ws + ws_alloc(off, (size_t)NB*RC*QP*2));
    __bf16* w_outb = (__bf16*)(ws + ws_alloc(off, (size_t)RC*RC*2));

    auto ceil_div = [](size_t a, size_t b){ return (a + b - 1) / b; };
    auto cast = [&](const float* src, __bf16* dst, size_t n) {
        cast_bf16<<<dim3((unsigned)ceil_div(n,256)), dim3(256), 0, stream>>>(src, dst, n);
    };

    // 1. bf16 casts + weight packing
    cast(x,     xb,     (size_t)NB*IC*QP);
    cast(h0,    h0b,    (size_t)NB*RC*QP);
    cast(w_in,  w_inb,  (size_t)RC*IC);
    cast(wg_a,  wgab,   (size_t)1024*RC);
    cast(w_out, w_outb, (size_t)RC*RC);
    {
        size_t n = (size_t)MSAME*KSM;
        pack_wsame<<<dim3((unsigned)ceil_div(n,256)), dim3(256), 0, stream>>>(wq_x, wq_h, wg_x, wg_h, Wsame);
        n = (size_t)MVAL*KSM;
        pack_wvalid<<<dim3((unsigned)ceil_div(n,256)), dim3(256), 0, stream>>>(wk_x, wk_h, wv_x, wv_h, Wvalid);
    }

    // 2. xin = w_in * x + b_in   (bf16 out)
    {
        BDense bl{xb, (long)IC*QP, QP};
        gemm_wmma<BDense><<<dim3(QP/64, RC/64, NB), dim3(128), 0, stream>>>(
            w_inb, bl, nullptr, 0, b_in, nullptr, xinb, (long)RC*QP, RC, QP, IC);
    }
    // 3. SAME conv: rows 0..255 -> q, rows 256..1279 -> gates partial
    {
        BConvSame bl{xinb, h0b};
        gemm_wmma<BConvSame><<<dim3(QP/64, MSAME/64, NB), dim3(128), 0, stream>>>(
            Wsame, bl, nullptr, 0, nullptr, qg, nullptr, (long)MSAME*QP, MSAME, QP, KSM);
    }
    // 4. VALID conv: rows 0..255 -> k, rows 256..511 -> v (bias bv added later)
    {
        BConvValid bl{xinb, h0b};
        gemm_wmma<BConvValid><<<dim3((unsigned)ceil_div(DV,64), MVAL/64, NB), dim3(128), 0, stream>>>(
            Wvalid, bl, nullptr, 0, nullptr, kvf, nullptr, (long)MVAL*DV, MVAL, DV, KSM);
    }
    // 5. attention input layouts (bf16, per batch*head)
    {
        size_t n = (size_t)NB*8*QP*32;
        make_qT<<<dim3((unsigned)ceil_div(n,256)), dim3(256), 0, stream>>>(qg, qTb);
        n = (size_t)NB*8*32*512;
        make_kv<<<dim3((unsigned)ceil_div(n,256)), dim3(256), 0, stream>>>(kvf, bv, kTb, vbb);
    }
    // 6. fused scores -> softmax -> a   (writes ab bf16)
    attn_fused<<<dim3(QP/16, NB*8), dim3(128), 0, stream>>>(qTb, kTb, vbb, ab);

    // 7. gates = gatesPartial + bg + wg_a * a   (in-place on qg gate rows)
    {
        BDense bl{ab, (long)AC*QP, QP};
        float* gates = qg + (size_t)RC*QP;
        gemm_wmma<BDense><<<dim3(QP/64, 1024/64, NB), dim3(128), 0, stream>>>(
            wgab, bl, gates, (long)MSAME*QP, bg, gates, nullptr, (long)MSAME*QP, 1024, QP, RC);
    }
    // 8. LSTM pointwise -> hb (bf16)
    {
        size_t n = (size_t)NB*RC*QP;
        lstm_pw<<<dim3((unsigned)ceil_div(n,256)), dim3(256), 0, stream>>>(qg, c0, hb);
    }
    // 9. out = w_out * h + b_out  (f32 to d_out)
    {
        BDense bl{hb, (long)RC*QP, QP};
        gemm_wmma<BDense><<<dim3(QP/64, RC/64, NB), dim3(128), 0, stream>>>(
            w_outb, bl, nullptr, 0, b_out, out, nullptr, (long)RC*QP, RC, QP, RC);
    }
}
